// Model_21981642621237
// MI455X (gfx1250) — matrix operand order repaired and checked by reading, not compiled, not run on hardware
//
#include <hip/hip_runtime.h>

typedef _Float16 v16h __attribute__((ext_vector_type(16)));
typedef float    v8f  __attribute__((ext_vector_type(8)));
typedef __fp16   h2   __attribute__((ext_vector_type(2)));   // matches cvt_pkrtz return type

#define NL 11

struct Params {
  const float* x;
  float*       out;
  const float* W[NL];
  const float* b[NL];
  void*        ws;     // packed per-lane A images: [NL][32] x 32B
  int npairs;          // number of 64-row pair-tiles
};

union HReg {
  v16h v;
  h2 pair[8];
  _Float16 e[16];
};

// Hidden activations carried as t = tanh(y); the affine sigmoid map
// 0.5*t + 0.5 is folded into the NEXT layer's weights/bias, so the only
// per-element work between layers is tanh + pack.
__device__ __forceinline__ float tanh_act(float y) {
#if __has_builtin(__builtin_amdgcn_tanhf)
  return __builtin_amdgcn_tanhf(y);
#else
  // tanh(y) = 2*sigmoid(2y) - 1 = 2/(1+exp2(-2*log2(e)*y)) - 1
  float e = __builtin_amdgcn_exp2f(y * -2.8853900817779268f);
  return __builtin_fmaf(2.0f, __builtin_amdgcn_rcpf(1.0f + e), -1.0f);
#endif
}

// ---- one-wave kernel: build the block-diagonal dual-tile A images (f16) with
// ---- bias folded in as an augmented column at K=8 (P rows) / K=24 (Q rows).
// Layer 0:      A = 0.5*W,  bias col = 0.5*b                    (input is x)
// Layers 1..10: A = 0.25*W, bias col = 0.5*b + 0.25*rowsum(W)   (input is tanh)
// A element map:  lanes<16 : e[k]   = (M=lane,    K=k)     -> P weights
//                            e[8]   = (M=lane,    K=8)     -> P bias (lanes 0..fout-1)
//                 lanes>=16: e[k]   = (M=lane-16, K=16+k)  -> Q weights (lanes 24..24+fout-1)
//                            e[8]   = (M=lane-16, K=24)    -> Q bias (lanes 24..24+fout-1)
__global__ void pack_weights(Params p) {
  constexpr int FIN[NL]  = {8, 6, 6, 6, 6, 6, 6, 4, 4, 4, 4};
  constexpr int FOUT[NL] = {6, 6, 6, 6, 6, 6, 4, 4, 4, 4, 1};
  const int lane = threadIdx.x & 31;
  const bool lo  = lane < 16;
#pragma unroll
  for (int l = 0; l < NL; ++l) {
    const int fin = FIN[l], fout = FOUT[l];
    const float* W = p.W[l];
    const float* b = p.b[l];
    const float ws = (l == 0) ? 0.5f : 0.25f;
    HReg a;
#pragma unroll
    for (int k = 0; k < 8; ++k) {
      float vLo = (k < fin && lo && lane < fout)
                      ? ws * W[lane * fin + k]
                      : ((k < fin && lane >= 24 && (lane - 24) < fout)
                             ? ws * W[(lane - 24) * fin + k] : 0.0f);
      float vHi = 0.0f;
      if (k == 0 && lo && lane < fout) {                // P-rows bias column (K=8)
        int m = lane;
        float s = 0.5f * b[m];
        if (l > 0) {
          float rs = 0.0f;
#pragma unroll
          for (int kk = 0; kk < 8; ++kk) if (kk < fin) rs += W[m * fin + kk];
          s += 0.25f * rs;
        }
        vHi = s;
      }
      if (k == 0 && lane >= 24 && (lane - 24) < fout) { // Q-rows bias column (K=24)
        int m = lane - 24;
        float s = 0.5f * b[m];
        if (l > 0) {
          float rs = 0.0f;
#pragma unroll
          for (int kk = 0; kk < 8; ++kk) if (kk < fin) rs += W[m * fin + kk];
          s += 0.25f * rs;
        }
        vHi = s;
      }
      a.e[k]     = (_Float16)vLo;
      a.e[8 + k] = (_Float16)vHi;
    }
    ((v16h*)p.ws)[l * 32 + lane] = a.v;
  }
}

__global__ __launch_bounds__(256) void mlp11_wmma(Params p) {
  constexpr int FOUT[NL] = {6, 6, 6, 6, 6, 6, 4, 4, 4, 4, 1};

  const int lane = threadIdx.x & 31;                               // wave32
  const int wid  = (blockIdx.x * blockDim.x + threadIdx.x) >> 5;
  const int nw   = (gridDim.x * blockDim.x) >> 5;

  // Load pre-packed per-lane A images (11 x 32B, L2-resident).
  HReg A[NL];
#pragma unroll
  for (int l = 0; l < NL; ++l)
    A[l].v = ((const v16h*)p.ws)[l * 32 + lane];

  v8f zc;                                               // zero C accumulator (shared)
#pragma unroll
  for (int m = 0; m < 8; ++m) zc[m] = 0.0f;

  // ---- wave-stride loop: two independent 32-row tiles per iteration; each
  // chain's tanh/pack VALU fills the other's WMMA->VALU hazard slots.
  for (int t = wid; t < p.npairs; t += nw) {
    const int rowA = t * 64 + lane;
    const int rowB = rowA + 32;

    if (t + nw < p.npairs)  // prefetch next pair-tile's x (global_prefetch_b8)
      __builtin_prefetch(&p.x[(size_t)(t + nw) * 512 + (size_t)lane * 16], 0, 1);

    const float4 a0 = ((const float4*)p.x)[rowA * 2 + 0];
    const float4 a1 = ((const float4*)p.x)[rowA * 2 + 1];
    const float4 b0 = ((const float4*)p.x)[rowB * 2 + 0];
    const float4 b1 = ((const float4*)p.x)[rowB * 2 + 1];

    // B matrix: element k = feature k of this lane's batch column.
    // Lanes 0-15 feed K=0..7 (tile P), lanes 16-31 feed K=16..23 (tile Q).
    // Element 8 = constant 1.0 -> multiplies the bias column of A (K=8 / K=24).
    // Pads are initialized ONCE; inside the layer loop only the live data pairs
    // are rewritten in place. Stale values at dead K slots are always finite
    // (old tanh outputs or x) and are multiplied by zero columns of A.
    HReg bmA, bmB;
    bmA.e[8] = (_Float16)1.0f;
    bmB.e[8] = (_Float16)1.0f;
#pragma unroll
    for (int e = 9; e < 16; ++e) { bmA.e[e] = (_Float16)0.0f; bmB.e[e] = (_Float16)0.0f; }
    bmA.pair[0] = __builtin_amdgcn_cvt_pkrtz(a0.x, a0.y);
    bmA.pair[1] = __builtin_amdgcn_cvt_pkrtz(a0.z, a0.w);
    bmA.pair[2] = __builtin_amdgcn_cvt_pkrtz(a1.x, a1.y);
    bmA.pair[3] = __builtin_amdgcn_cvt_pkrtz(a1.z, a1.w);
    bmB.pair[0] = __builtin_amdgcn_cvt_pkrtz(b0.x, b0.y);
    bmB.pair[1] = __builtin_amdgcn_cvt_pkrtz(b0.z, b0.w);
    bmB.pair[2] = __builtin_amdgcn_cvt_pkrtz(b1.x, b1.y);
    bmB.pair[3] = __builtin_amdgcn_cvt_pkrtz(b1.z, b1.w);

    float resA = 0.0f, resB = 0.0f;
#pragma unroll
    for (int l = 0; l < NL; ++l) {
      // D[m, n] (lanes 0-15) = P pre-activation feature m; D[m+8, n] (lanes 16-31) = Q.
      v8f dA = __builtin_amdgcn_wmma_f32_16x16x32_f16(
          false, A[l].v, false, bmA.v, (short)0, zc, false, false);
      v8f dB = __builtin_amdgcn_wmma_f32_16x16x32_f16(
          false, A[l].v, false, bmB.v, (short)0, zc, false, false);

      if (l == NL - 1) {
        // Final layer: apply the full sigmoid = 0.5*tanh + 0.5 once.
        resA = __builtin_fmaf(0.5f, tanh_act(dA[0]), 0.5f);
        resB = __builtin_fmaf(0.5f, tanh_act(dB[0]), 0.5f);
      } else {
#pragma unroll
        for (int j = 0; 2 * j < FOUT[l]; ++j) {   // all hidden fout are even
          bmA.pair[j] = __builtin_amdgcn_cvt_pkrtz(tanh_act(dA[2 * j]), tanh_act(dA[2 * j + 1]));
          bmB.pair[j] = __builtin_amdgcn_cvt_pkrtz(tanh_act(dB[2 * j]), tanh_act(dB[2 * j + 1]));
        }
      }
    }
    p.out[rowA] = resA;
    p.out[rowB] = resB;
  }
}

extern "C" void kernel_launch(void* const* d_in, const int* in_sizes, int n_in,
                              void* d_out, int out_size, void* d_ws, size_t ws_size,
                              hipStream_t stream) {
  Params p;
  p.x   = (const float*)d_in[0];
  p.out = (float*)d_out;
  for (int i = 0; i < NL; ++i) {
    p.W[i] = (const float*)d_in[1 + 2 * i];
    p.b[i] = (const float*)d_in[2 + 2 * i];
  }
  p.ws = d_ws;                         // needs NL*32*32 = 11264 bytes
  const int nrows = in_sizes[0] / 8;   // x is [B, 8]
  p.npairs = nrows / 64;               // 64 batch rows (2 WMMA tiles) per wave iteration

  hipLaunchKernelGGL(pack_weights, dim3(1), dim3(32), 0, stream, p);
  hipLaunchKernelGGL(mlp11_wmma, dim3(2048), dim3(256), 0, stream, p);
}
